// PLE_27350351740993
// MI455X (gfx1250) — matrix operand order repaired and checked by reading, not compiled, not run on hardware
//
#include <hip/hip_runtime.h>
#include <stdint.h>

// PLE / CGC multi-task MoE for MI455X (gfx1250, wave32, WMMA + TDM).
// bf16 operands + f32 accumulate via v_wmma_f32_16x16x32_bf16.
// Global->LDS staging done by the Tensor Data Mover (tensor_load_to_lds),
// with D#-descriptor padding reproducing the conflict-free 80B LDS pitch.

typedef __bf16 bf16_t;
typedef __attribute__((ext_vector_type(16))) __bf16 v16bf;
typedef __attribute__((ext_vector_type(8)))  float  v8f;
typedef __attribute__((ext_vector_type(4)))  unsigned int u32x4;
typedef __attribute__((ext_vector_type(8)))  unsigned int u32x8;

static constexpr int Bd = 16384, Dd = 1024, H1d = 512, H2d = 256;
static constexpr int Td = 3;

__device__ inline unsigned short f2bfbits(float f) {
  unsigned u = __builtin_bit_cast(unsigned, f);
  u += 0x7FFFu + ((u >> 16) & 1u);   // round-to-nearest-even
  return (unsigned short)(u >> 16);
}
__device__ inline bf16_t f2bf(float f) {
  unsigned short h = f2bfbits(f);
  return __builtin_bit_cast(bf16_t, h);
}
__device__ inline float bf2f(bf16_t v) {
  unsigned short u = __builtin_bit_cast(unsigned short, v);
  unsigned x = ((unsigned)u) << 16;
  return __builtin_bit_cast(float, x);
}

// ---------------- fp32 -> bf16 straight convert (vectorized) ----------------
__global__ __launch_bounds__(256) void k_cvt_f32_bf16(const float4* __restrict__ in,
                                                      uint2* __restrict__ out, int n4) {
  for (int i = blockIdx.x * blockDim.x + threadIdx.x; i < n4; i += gridDim.x * blockDim.x) {
    float4 v = in[i];
    unsigned lo = (unsigned)f2bfbits(v.x) | ((unsigned)f2bfbits(v.y) << 16);
    unsigned hi = (unsigned)f2bfbits(v.z) | ((unsigned)f2bfbits(v.w) << 16);
    out[i] = make_uint2(lo, hi);
  }
}

// ------------- fp32 [M][R][C] -> bf16 [M][C][R] transpose-convert -----------
__global__ __launch_bounds__(256) void k_transpose_cvt(const float* __restrict__ in,
                                                       bf16_t* __restrict__ out,
                                                       int R, int C) {
  __shared__ float tile[32][33];
  const int m = blockIdx.z;
  const float* src = in + (size_t)m * R * C;
  bf16_t* dst = out + (size_t)m * R * C;
  const int c0 = blockIdx.x * 32, r0 = blockIdx.y * 32;
  const int tx = threadIdx.x, ty = threadIdx.y;
#pragma unroll
  for (int j = 0; j < 32; j += 8)
    tile[ty + j][tx] = src[(size_t)(r0 + ty + j) * C + c0 + tx];
  __syncthreads();
#pragma unroll
  for (int j = 0; j < 32; j += 8)
    dst[(size_t)(c0 + ty + j) * R + r0 + tx] = f2bf(tile[tx][ty + j]);
}

// ---------------------- Tensor Data Mover helpers ---------------------------
// D# group1 for a 2D 128x32-element (rows x halves) bf16 tile:
//   data_size=2B (code 1), pad_enable=1, pad_interval=3 (pad every 64B = one
//   32-half row), pad_amount=3 (16B) -> LDS row pitch 80B == PITCH=40 halves.
//   tensor_dim0 = tile_dim0 = 32, tensor_dim1 = tile_dim1 = 128,
//   tensor_dim0_stride = row stride in elements.
__device__ inline u32x8 tdm_desc_g1(unsigned stride_elems) {
  u32x8 g = {};
  g[0] = (1u << 16) | (1u << 20) | (3u << 22) | (3u << 25);  // 0x06D10000
  g[1] = (32u << 16);    // tensor_dim0[15:0] in bits[63:48]
  g[2] = (128u << 16);   // tensor_dim1[15:0] in bits[95:80]
  g[3] = (32u << 16);    // tile_dim0 in bits[127:112]
  g[4] = 128u;           // tile_dim1 in bits[143:128]
  g[5] = stride_elems;   // tensor_dim0_stride[31:0] (bits[191:160])
  g[6] = 0u;
  g[7] = 0u;
  return g;
}

__device__ inline void tdm_load_2d(unsigned lds_byte_addr, const void* gaddr, u32x8 g1) {
  unsigned long long ga = (unsigned long long)gaddr;
  u32x4 g0;
  g0[0] = 1u;                          // count=1, user descriptor
  g0[1] = lds_byte_addr;               // lds_addr (bits[63:32])
  g0[2] = (unsigned)ga;                // global_addr[31:0]
  g0[3] = (unsigned)((ga >> 32) & 0x01FFFFFFu) | 0x80000000u;  // addr[56:32] | type=2
  asm volatile("tensor_load_to_lds %0, %1" :: "s"(g0), "s"(g1) : "memory");
}

// ---------------------------- WMMA GEMM ------------------------------------
// C[b, ge*NPE + n] = relu( A[b, ge*aColStride + k] * Bt[ge][n][k] + bias[ge][n] )
// Block tile 128x128, BK=32, 256 threads = 8 waves, wave tile 32(M)x64(N).
union Frag {
  v16bf v;
  uint4 q[2];
};

__global__ __launch_bounds__(256) void k_gemm_wmma(
    const bf16_t* __restrict__ A,
    const bf16_t* __restrict__ BtSh,   // [2][NPE][K]
    const bf16_t* __restrict__ BtSp,   // [6][NPE][K]
    const float* __restrict__ biasSh,  // [2][NPE]
    const float* __restrict__ biasSp,  // [6][NPE]
    bf16_t* __restrict__ C,            // [B, ldc]
    int K, int NPE, int lda, int ldc, int aColStride) {
  constexpr int PITCH = 40;  // halves; 80B rows (TDM pad) keep 16B align, no conflicts
  __shared__ bf16_t As[2][128 * PITCH];
  __shared__ bf16_t Bs[2][128 * PITCH];
  constexpr unsigned BUFB = 128 * PITCH * 2;  // bytes per buffer (10240)

  const int tid = threadIdx.x;
  const int lane = tid & 31, wave = tid >> 5;
  const int wm = wave & 3, wn = wave >> 2;

  const int nbpe = NPE >> 7;                       // 128-wide N blocks per expert
  const int ge = blockIdx.x / nbpe;                // global expert id 0..7
  const int n0 = (blockIdx.x - ge * nbpe) * 128;   // N offset inside expert
  const int m0 = blockIdx.y * 128;

  const bf16_t* Bt = (ge < 2) ? (BtSh + (size_t)ge * NPE * K)
                              : (BtSp + (size_t)(ge - 2) * NPE * K);
  const float* bias = (ge < 2) ? (biasSh + ge * NPE) : (biasSp + (ge - 2) * NPE);

  const bf16_t* Ab = A + (size_t)m0 * lda + (size_t)ge * aColStride;
  const bf16_t* Bb = Bt + (size_t)n0 * K;

  // LDS byte addresses (generic LDS pointer low 32 bits == LDS offset).
  const unsigned asb = (unsigned)(size_t)(void*)&As[0][0];
  const unsigned bsb = (unsigned)(size_t)(void*)&Bs[0][0];

  const u32x8 g1a = tdm_desc_g1((unsigned)lda);
  const u32x8 g1b = tdm_desc_g1((unsigned)K);
  const bool issuer = (wave == 0);

  auto issue = [&](int kt, int buf) {
    tdm_load_2d(asb + (unsigned)buf * BUFB, Ab + kt * 32, g1a);
    tdm_load_2d(bsb + (unsigned)buf * BUFB, Bb + kt * 32, g1b);
  };

  v8f acc[2][4];
#pragma unroll
  for (int i = 0; i < 2; ++i)
#pragma unroll
    for (int j = 0; j < 4; ++j) acc[i][j] = (v8f){0.f, 0.f, 0.f, 0.f, 0.f, 0.f, 0.f, 0.f};

  if (issuer) issue(0, 0);

  const int KT = K >> 5;
  for (int kt = 0; kt < KT; ++kt) {
    const int buf = kt & 1;
    if (issuer) __builtin_amdgcn_s_wait_tensorcnt(0);  // tile kt landed in LDS
    __syncthreads();                                    // visible to all waves;
                                                        // all done reading buf^1
    if (issuer && (kt + 1 < KT)) issue(kt + 1, 1 - buf);  // DMA overlaps compute

    // Fragment loads from LDS per documented wave32 WMMA layouts.
    const bf16_t* as = As[buf];
    const bf16_t* bs = Bs[buf];
    const int mr = wm * 32 + (lane & 15);
    const int nr = wn * 64 + (lane & 15);
    const int kA = (lane >> 4) * 8;    // A: lo=K[kA..kA+7], hi=K[kA+16..kA+23]
    const int kB = (lane >> 4) * 8;    // B: lo=K[kB..kB+7], hi=K[kB+16..kB+23] of column (lane&15)

    Frag af[2], bfr[4];
#pragma unroll
    for (int tm = 0; tm < 2; ++tm) {
      const bf16_t* p = as + (mr + tm * 16) * PITCH + kA;
      af[tm].q[0] = *(const uint4*)p;
      af[tm].q[1] = *(const uint4*)(p + 16);
    }
#pragma unroll
    for (int tn = 0; tn < 4; ++tn) {
      const bf16_t* p = bs + (nr + tn * 16) * PITCH + kB;
      bfr[tn].q[0] = *(const uint4*)p;
      bfr[tn].q[1] = *(const uint4*)(p + 16);
    }
#pragma unroll
    for (int tm = 0; tm < 2; ++tm)
#pragma unroll
      for (int tn = 0; tn < 4; ++tn)
        acc[tm][tn] = __builtin_amdgcn_wmma_f32_16x16x32_bf16(
            false, af[tm].v, false, bfr[tn].v, (short)0, acc[tm][tn], false, false);
  }

  // epilogue: bias + relu, bf16 store
#pragma unroll
  for (int tm = 0; tm < 2; ++tm) {
    const int r0 = m0 + wm * 32 + tm * 16 + (lane >> 4) * 8;
#pragma unroll
    for (int tn = 0; tn < 4; ++tn) {
      const int ce = n0 + wn * 64 + tn * 16 + (lane & 15);
      const float bv = bias[ce];
      const size_t gc = (size_t)ge * NPE + ce;
#pragma unroll
      for (int v = 0; v < 8; ++v) {
        float r = acc[tm][tn][v] + bv;
        r = r > 0.f ? r : 0.f;
        C[(size_t)(r0 + v) * ldc + gc] = f2bf(r);
      }
    }
  }
}

// -------------------------- gates: dot + softmax ----------------------------
__global__ __launch_bounds__(256) void k_gate(const float* __restrict__ x,
                                              const float* __restrict__ gW,   // [3][1024][4]
                                              const float* __restrict__ gb,   // [3][4]
                                              float* __restrict__ gates) {    // [B][12]
  const int lane = threadIdx.x & 31, wave = threadIdx.x >> 5;
  const int b = blockIdx.x * 8 + wave;
  const float* xr = x + (size_t)b * Dd;
  float acc[12];
#pragma unroll
  for (int i = 0; i < 12; ++i) acc[i] = 0.f;
  for (int d = lane; d < Dd; d += 32) {
    const float xv = xr[d];
#pragma unroll
    for (int t = 0; t < 3; ++t)
#pragma unroll
      for (int j = 0; j < 4; ++j) acc[t * 4 + j] += xv * gW[t * Dd * 4 + d * 4 + j];
  }
#pragma unroll
  for (int i = 0; i < 12; ++i)
    for (int off = 16; off > 0; off >>= 1) acc[i] += __shfl_down(acc[i], off, 32);
  if (lane == 0) {
    float* o = gates + (size_t)b * 12;
#pragma unroll
    for (int t = 0; t < 3; ++t) {
      float l0 = acc[t * 4 + 0] + gb[t * 4 + 0];
      float l1 = acc[t * 4 + 1] + gb[t * 4 + 1];
      float l2 = acc[t * 4 + 2] + gb[t * 4 + 2];
      float l3 = acc[t * 4 + 3] + gb[t * 4 + 3];
      float mx = fmaxf(fmaxf(l0, l1), fmaxf(l2, l3));
      float e0 = __expf(l0 - mx), e1 = __expf(l1 - mx), e2 = __expf(l2 - mx), e3 = __expf(l3 - mx);
      float inv = 1.f / (e0 + e1 + e2 + e3);
      o[t * 4 + 0] = e0 * inv;
      o[t * 4 + 1] = e1 * inv;
      o[t * 4 + 2] = e2 * inv;
      o[t * 4 + 3] = e3 * inv;
    }
  }
}

// ------------------ combine: gate-weighted expert mixture -------------------
__global__ __launch_bounds__(256) void k_combine(const bf16_t* __restrict__ out2,  // [B][8*256]
                                                 const float* __restrict__ gates,  // [B][12]
                                                 float* __restrict__ out) {        // [B][3][256]
  const int bt = blockIdx.x;
  const int b = bt / 3, t = bt - b * 3;
  const int h = threadIdx.x;
  const float* g = gates + (size_t)b * 12 + t * 4;
  const bf16_t* o2 = out2 + (size_t)b * (8 * H2d);
  float r = g[0] * bf2f(o2[0 * H2d + h]) + g[1] * bf2f(o2[1 * H2d + h]) +
            g[2] * bf2f(o2[(2 + 2 * t) * H2d + h]) + g[3] * bf2f(o2[(3 + 2 * t) * H2d + h]);
  out[(size_t)bt * H2d + h] = r;
}

// ----------------------------------------------------------------------------
extern "C" void kernel_launch(void* const* d_in, const int* in_sizes, int n_in,
                              void* d_out, int out_size, void* d_ws, size_t ws_size,
                              hipStream_t stream) {
  (void)in_sizes; (void)n_in; (void)out_size; (void)ws_size;
  const float* x    = (const float*)d_in[0];
  const float* shW1 = (const float*)d_in[1];
  const float* shB1 = (const float*)d_in[2];
  const float* shW2 = (const float*)d_in[3];
  const float* shB2 = (const float*)d_in[4];
  const float* spW1 = (const float*)d_in[5];
  const float* spB1 = (const float*)d_in[6];
  const float* spW2 = (const float*)d_in[7];
  const float* spB2 = (const float*)d_in[8];
  const float* gW   = (const float*)d_in[9];
  const float* gb   = (const float*)d_in[10];
  float* out = (float*)d_out;

  char* ws = (char*)d_ws;
  size_t off = 0;
  auto walloc = [&](size_t bytes) -> void* {
    void* p = ws + off;
    off += (bytes + 255) & ~(size_t)255;
    return p;
  };
  bf16_t* x_bf  = (bf16_t*)walloc((size_t)Bd * Dd * 2);        // 33.5 MB
  bf16_t* w1tsh = (bf16_t*)walloc((size_t)2 * H1d * Dd * 2);   // [2][512][1024]
  bf16_t* w1tsp = (bf16_t*)walloc((size_t)6 * H1d * Dd * 2);   // [6][512][1024]
  bf16_t* w2tsh = (bf16_t*)walloc((size_t)2 * H2d * H1d * 2);  // [2][256][512]
  bf16_t* w2tsp = (bf16_t*)walloc((size_t)6 * H2d * H1d * 2);  // [6][256][512]
  bf16_t* h1    = (bf16_t*)walloc((size_t)Bd * 8 * H1d * 2);   // 134 MB
  bf16_t* o2    = (bf16_t*)walloc((size_t)Bd * 8 * H2d * 2);   // 67 MB
  float*  gates = (float*)walloc((size_t)Bd * 12 * 4);         // 0.8 MB

  // 1) conversions (x straight; weights transposed to [N][K] bf16)
  k_cvt_f32_bf16<<<4096, 256, 0, stream>>>((const float4*)x, (uint2*)x_bf, Bd * Dd / 4);
  k_transpose_cvt<<<dim3(H1d / 32, Dd / 32, 2), dim3(32, 8), 0, stream>>>(shW1, w1tsh, Dd, H1d);
  k_transpose_cvt<<<dim3(H1d / 32, Dd / 32, 6), dim3(32, 8), 0, stream>>>(spW1, w1tsp, Dd, H1d);
  k_transpose_cvt<<<dim3(H2d / 32, H1d / 32, 2), dim3(32, 8), 0, stream>>>(shW2, w2tsh, H1d, H2d);
  k_transpose_cvt<<<dim3(H2d / 32, H1d / 32, 6), dim3(32, 8), 0, stream>>>(spW2, w2tsp, H1d, H2d);

  // 2) gates (independent of expert path)
  k_gate<<<Bd / 8, 256, 0, stream>>>(x, gW, gb, gates);

  // 3) layer-1: A=[B,1024] bf16, fused N = 8 experts x 512 -> h1 [B,4096] bf16
  k_gemm_wmma<<<dim3(8 * (H1d / 128), Bd / 128), 256, 0, stream>>>(
      x_bf, w1tsh, w1tsp, shB1, spB1, h1, Dd, H1d, Dd, 8 * H1d, 0);

  // 4) layer-2: per-expert K=512 slice of h1 -> o2 [B,2048] bf16
  k_gemm_wmma<<<dim3(8 * (H2d / 128), Bd / 128), 256, 0, stream>>>(
      h1, w2tsh, w2tsp, shB2, spB2, o2, H1d, H2d, 8 * H1d, 8 * H2d, H1d);

  // 5) gate-weighted combine -> out [B,3,256] fp32
  k_combine<<<Bd * Td, 256, 0, stream>>>(o2, gates, out);
}
